// Attention_14044543058524
// MI455X (gfx1250) — compile-verified, matrix operand order checked by reading, not run on hardware
//
#include <hip/hip_runtime.h>
#include <math.h>

#define D_DIM 4
#define B_DIM 16384
#define H_DIM 1024
#define BM 128
#define BN 128
#define KS 32
#define LPAD 8   // LDS row padding (bf16 elems) to spread banks

typedef __attribute__((ext_vector_type(16))) __bf16 v16bf;
typedef __attribute__((ext_vector_type(8)))  __bf16 v8bf;
typedef __attribute__((ext_vector_type(8)))  float  v8f;

// Build a 16x32 bf16 A/B fragment per the CDNA5 16-bit operand layout:
// lane L (0..15): row = L, K = {kh..kh+7, kh+16..kh+23} with kh = (L>=16)*8.
__device__ __forceinline__ v16bf load_frag(const __bf16* p) {
  const v8bf lo = *reinterpret_cast<const v8bf*>(p);
  const v8bf hi = *reinterpret_cast<const v8bf*>(p + 16);
  return __builtin_shufflevector(lo, hi, 0,1,2,3,4,5,6,7,8,9,10,11,12,13,14,15);
}

// CDNA5 async global->LDS copy (ASYNCcnt path). vdst = LDS byte address,
// vaddr = 64-bit global address (GV mode, saddr = off).
__device__ __forceinline__ void async_load_b128(unsigned lds_off, const void* gptr) {
  asm volatile("global_load_async_to_lds_b128 %0, %1, off"
               :: "v"(lds_off), "v"(gptr) : "memory");
}
__device__ __forceinline__ void wait_asynccnt0() {
  asm volatile("s_wait_asynccnt 0x0" ::: "memory");
}

// ---------------------------------------------------------------------------
// K0a: zero the 16-float score accumulator (must be re-zeroed every launch)
// ---------------------------------------------------------------------------
__global__ void zero16_kernel(float* __restrict__ p) {
  if (threadIdx.x < 16) p[threadIdx.x] = 0.0f;
}

// ---------------------------------------------------------------------------
// K0b: pre-convert weights f32 -> bf16 (8 MB; L2-resident for the GEMM)
// 8 elems/thread; grid*block*8 == 4*1024*1024
// ---------------------------------------------------------------------------
__global__ __launch_bounds__(256) void wconv_kernel(const float* __restrict__ w,
                                                    __bf16* __restrict__ wb) {
  const size_t i = ((size_t)blockIdx.x * 256 + threadIdx.x) * 8;
  const float4 f0 = *reinterpret_cast<const float4*>(w + i);
  const float4 f1 = *reinterpret_cast<const float4*>(w + i + 4);
  __bf16 t[8];
  t[0] = (__bf16)f0.x; t[1] = (__bf16)f0.y; t[2] = (__bf16)f0.z; t[3] = (__bf16)f0.w;
  t[4] = (__bf16)f1.x; t[5] = (__bf16)f1.y; t[6] = (__bf16)f1.z; t[7] = (__bf16)f1.w;
  *reinterpret_cast<uint4*>(wb + i) = *reinterpret_cast<const uint4*>(t);
}

// ---------------------------------------------------------------------------
// K1: Y_d = X_d @ W_d^T  (bf16 WMMA, f32 accumulate, Y stored bf16 in ws)
// X staged f32->bf16 through VGPRs; W staged bf16 via async global->LDS.
// grid = (H/BN, B/BM, D), block = 256 (8 waves)
// ---------------------------------------------------------------------------
__global__ __launch_bounds__(256) void gemm_y_kernel(const float* __restrict__ feats,
                                                     const __bf16* __restrict__ wbf,
                                                     __bf16* __restrict__ Y) {
  __shared__ __bf16 Xs[BM][KS + LPAD];
  __shared__ __bf16 Ws[BN][KS + LPAD];

  const int n0 = blockIdx.x * BN;
  const int m0 = blockIdx.y * BM;
  const int d  = blockIdx.z;

  const float*  __restrict__ X  = feats + (size_t)d * B_DIM * H_DIM;
  const __bf16* __restrict__ Wb = wbf   + (size_t)d * H_DIM * H_DIM;
  __bf16* __restrict__ Yd = Y + (size_t)d * B_DIM * H_DIM;

  const int tid  = threadIdx.x;
  const int wave = tid >> 5;
  const int lane = tid & 31;
  const int mw   = (wave & 3) * 32;   // 4 m-waves cover 128 rows
  const int nw   = (wave >> 2) * 64;  // 2 n-waves cover 128 cols
  const int frow = lane & 15;
  const int kh   = (lane >> 4) * 8;

  // X staging: thread -> (row, 16-elem half-row)
  const int srow = tid >> 1;
  const int scol = (tid & 1) * 16;

  // W async staging: 512 16-byte chunks; thread handles chunks tid, tid+256.
  // chunk c -> row = c>>2, quarter q = c&3 (q*8 bf16 within the K-slab)
  const unsigned ws_base = (unsigned)(uintptr_t)&Ws[0][0];
  const int wrow0 = tid >> 2,           wq0 = tid & 3;
  const int wrow1 = (tid + 256) >> 2,   wq1 = tid & 3;
  const unsigned wdst0 = ws_base + (unsigned)(wrow0 * (KS + LPAD) * 2 + wq0 * 16);
  const unsigned wdst1 = ws_base + (unsigned)(wrow1 * (KS + LPAD) * 2 + wq1 * 16);

  v8f acc[2][4];
  #pragma unroll
  for (int ms = 0; ms < 2; ++ms)
    #pragma unroll
    for (int ns = 0; ns < 4; ++ns)
      acc[ms][ns] = (v8f){0.f,0.f,0.f,0.f,0.f,0.f,0.f,0.f};

  for (int k0 = 0; k0 < H_DIM; k0 += KS) {
    // ---- W slab: async global->LDS (bf16, pure byte copy) ----
    async_load_b128(wdst0, Wb + (size_t)(n0 + wrow0) * H_DIM + k0 + wq0 * 8);
    async_load_b128(wdst1, Wb + (size_t)(n0 + wrow1) * H_DIM + k0 + wq1 * 8);

    // ---- X slab: f32 load -> v_cvt_pk_bf16_f32 -> ds_store ----
    {
      const float4* xsrc = reinterpret_cast<const float4*>(
          X + (size_t)(m0 + srow) * H_DIM + k0 + scol);
      __bf16 tx[16];
      #pragma unroll
      for (int v = 0; v < 4; ++v) {
        const float4 fx = xsrc[v];
        tx[4*v+0] = (__bf16)fx.x; tx[4*v+1] = (__bf16)fx.y;
        tx[4*v+2] = (__bf16)fx.z; tx[4*v+3] = (__bf16)fx.w;
      }
      uint4* dx = reinterpret_cast<uint4*>(&Xs[srow][scol]);
      dx[0] = reinterpret_cast<const uint4*>(tx)[0];
      dx[1] = reinterpret_cast<const uint4*>(tx)[1];
    }
    if (k0 + KS < H_DIM)  // pull next X K-slab toward L2 (global_prefetch_b8)
      __builtin_prefetch(X + (size_t)(m0 + srow) * H_DIM + k0 + KS + scol, 0, 1);

    wait_asynccnt0();     // this wave's async W chunks landed in LDS
    __syncthreads();      // all waves' X ds_stores + W async copies visible

    v16bf a[2];
    #pragma unroll
    for (int ms = 0; ms < 2; ++ms)
      a[ms] = load_frag(&Xs[mw + ms * 16 + frow][kh]);
    #pragma unroll
    for (int ns = 0; ns < 4; ++ns) {
      const v16bf b = load_frag(&Ws[nw + ns * 16 + frow][kh]);
      #pragma unroll
      for (int ms = 0; ms < 2; ++ms)
        acc[ms][ns] = __builtin_amdgcn_wmma_f32_16x16x32_bf16(
            false, a[ms], false, b, (short)0, acc[ms][ns], false, false);
    }
    __syncthreads();      // slab consumed; safe to overwrite next iteration
  }

  // ---- epilogue: C/D layout (VGPR r -> M = r + (lane>=16)*8, N = lane%16) ----
  const int col  = lane & 15;
  const int roff = (lane >> 4) * 8;
  #pragma unroll
  for (int ms = 0; ms < 2; ++ms)
    #pragma unroll
    for (int ns = 0; ns < 4; ++ns)
      #pragma unroll
      for (int r = 0; r < 8; ++r) {
        const int row = m0 + mw + ms * 16 + roff + r;
        const int cc  = n0 + nw + ns * 16 + col;
        Yd[(size_t)row * H_DIM + cc] = (__bf16)acc[ms][ns][r];
      }
}

// ---------------------------------------------------------------------------
// K2: per-b pairwise dots P_ij(b) = Y_i[b]·Y_j[b]  ->  normalized cosine
//     scores[i,j] += sum_b P_ij / (max(sqrt(P_ii),eps)*max(sqrt(P_jj),eps))
// block = 256 = 16 b-rows x 16 H-segments; grid = B/16
// ---------------------------------------------------------------------------
__global__ __launch_bounds__(256) void score_kernel(const __bf16* __restrict__ Y,
                                                    float* __restrict__ scoreAcc) {
  __shared__ float red[16][10][16];  // [b][pair][seg]
  __shared__ float pb[16][10];
  __shared__ float invn[16][4];

  const int tid = threadIdx.x;
  const int seg = tid & 15;
  const int bl  = tid >> 4;
  const size_t b  = (size_t)blockIdx.x * 16 + bl;
  const size_t rs = (size_t)B_DIM * H_DIM;
  const __bf16* base = Y + b * H_DIM + (size_t)seg * 64;

  float p[10];
  #pragma unroll
  for (int q = 0; q < 10; ++q) p[q] = 0.f;

  for (int j = 0; j < 64; j += 8) {
    float v[4][8];
    #pragma unroll
    for (int i = 0; i < 4; ++i) {
      const v8bf vb = *reinterpret_cast<const v8bf*>(base + (size_t)i * rs + j);
      #pragma unroll
      for (int e = 0; e < 8; ++e) v[i][e] = (float)vb[e];
    }
    int q = 0;
    #pragma unroll
    for (int i = 0; i < 4; ++i)
      #pragma unroll
      for (int jj = i; jj < 4; ++jj, ++q)
        #pragma unroll
        for (int e = 0; e < 8; ++e)
          p[q] += v[i][e] * v[jj][e];
  }

  #pragma unroll
  for (int q = 0; q < 10; ++q) red[bl][q][seg] = p[q];
  __syncthreads();

  if (tid < 160) {  // reduce over segments
    const int bb = tid / 10, q = tid % 10;
    float s = 0.f;
    #pragma unroll
    for (int g = 0; g < 16; ++g) s += red[bb][q][g];
    pb[bb][q] = s;
  }
  __syncthreads();

  if (tid < 16) {   // per-b inverse norms (diag pairs q(i,i) = {0,4,7,9})
    const int diag[4] = {0, 4, 7, 9};
    #pragma unroll
    for (int i = 0; i < 4; ++i) {
      const float nrm = sqrtf(pb[tid][diag[i]]);
      invn[tid][i] = 1.0f / fmaxf(nrm, 1e-12f);
    }
  }
  __syncthreads();

  if (tid < 16) {   // one thread per (i,j): sum over this block's 16 b's
    const int i = tid >> 2, j = tid & 3;
    const int ii = (i < j) ? i : j;
    const int jj = (i < j) ? j : i;
    const int q  = ii * 4 - (ii * (ii - 1)) / 2 + (jj - ii);  // upper-tri index
    float s = 0.f;
    #pragma unroll
    for (int bb = 0; bb < 16; ++bb)
      s += pb[bb][q] * invn[bb][i] * invn[bb][j];
    atomicAdd(&scoreAcc[tid], s);
  }
}

// ---------------------------------------------------------------------------
// K3: 4x4 softmax over rows (axis=1), scores scaled by 1/B
// ---------------------------------------------------------------------------
__global__ void softmax_kernel(const float* __restrict__ scoreAcc,
                               float* __restrict__ attn) {
  const int t = threadIdx.x;
  if (t < 4) {
    float r[4];
    #pragma unroll
    for (int c = 0; c < 4; ++c) r[c] = scoreAcc[t * 4 + c] * (1.0f / (float)B_DIM);
    float m = r[0];
    #pragma unroll
    for (int c = 1; c < 4; ++c) m = fmaxf(m, r[c]);
    float s = 0.f;
    #pragma unroll
    for (int c = 0; c < 4; ++c) { r[c] = expf(r[c] - m); s += r[c]; }
    const float inv = 1.0f / s;
    #pragma unroll
    for (int c = 0; c < 4; ++c) attn[t * 4 + c] = r[c] * inv;
  }
}

// ---------------------------------------------------------------------------
// K4: out[d] = sum_g attn[d,g] * feats[g]   (pure bandwidth, float4)
// ---------------------------------------------------------------------------
__global__ __launch_bounds__(256) void aggregate_kernel(const float* __restrict__ feats,
                                                        const float* __restrict__ attn,
                                                        float* __restrict__ out) {
  __shared__ float a[16];
  if (threadIdx.x < 16) a[threadIdx.x] = attn[threadIdx.x];
  __syncthreads();

  const size_t stride = (size_t)B_DIM * H_DIM;
  const size_t idx = ((size_t)blockIdx.x * blockDim.x + threadIdx.x) * 4;
  float4 f[4];
  #pragma unroll
  for (int g = 0; g < 4; ++g)
    f[g] = *reinterpret_cast<const float4*>(feats + (size_t)g * stride + idx);
  #pragma unroll
  for (int d = 0; d < 4; ++d) {
    float4 o;
    o.x = a[d*4+0]*f[0].x + a[d*4+1]*f[1].x + a[d*4+2]*f[2].x + a[d*4+3]*f[3].x;
    o.y = a[d*4+0]*f[0].y + a[d*4+1]*f[1].y + a[d*4+2]*f[2].y + a[d*4+3]*f[3].y;
    o.z = a[d*4+0]*f[0].z + a[d*4+1]*f[1].z + a[d*4+2]*f[2].z + a[d*4+3]*f[3].z;
    o.w = a[d*4+0]*f[0].w + a[d*4+1]*f[1].w + a[d*4+2]*f[2].w + a[d*4+3]*f[3].w;
    *reinterpret_cast<float4*>(out + (size_t)d * stride + idx) = o;
  }
}

// ---------------------------------------------------------------------------
extern "C" void kernel_launch(void* const* d_in, const int* in_sizes, int n_in,
                              void* d_out, int out_size, void* d_ws, size_t ws_size,
                              hipStream_t stream) {
  (void)in_sizes; (void)n_in; (void)out_size; (void)ws_size;
  const float* feats   = (const float*)d_in[0];   // [4,16384,1024] f32
  const float* weights = (const float*)d_in[1];   // [4,1024,1024]  f32
  float* out = (float*)d_out;                     // [4,16384,1024] f32

  char* ws = (char*)d_ws;
  float*  scoreAcc = (float*)ws;                          // 16 floats
  float*  attn     = (float*)(ws + 64);                   // 16 floats
  __bf16* Wb       = (__bf16*)(ws + 256);                 // bf16 weights, 8 MiB
  __bf16* Y        = (__bf16*)(ws + 256 + (size_t)D_DIM * H_DIM * H_DIM * 2); // 128 MiB

  zero16_kernel<<<1, 32, 0, stream>>>(scoreAcc);

  const int wconv_blocks = (D_DIM * H_DIM * H_DIM) / (256 * 8);  // 2048
  wconv_kernel<<<wconv_blocks, 256, 0, stream>>>(weights, Wb);

  dim3 g1(H_DIM / BN, B_DIM / BM, D_DIM);                 // (8, 128, 4)
  gemm_y_kernel<<<g1, 256, 0, stream>>>(feats, Wb, Y);

  score_kernel<<<B_DIM / 16, 256, 0, stream>>>(Y, scoreAcc);

  softmax_kernel<<<1, 32, 0, stream>>>(scoreAcc, attn);

  const int agg_blocks = (int)(((size_t)B_DIM * H_DIM / 4) / 256);  // 16384
  aggregate_kernel<<<agg_blocks, 256, 0, stream>>>(feats, attn, out);
}
